// HetGAT_76682346102820
// MI455X (gfx1250) — matrix operand order repaired and checked by reading, not compiled, not run on hardware
//
#include <hip/hip_runtime.h>
#include <cstddef>

// ---------------------------------------------------------------------------
// Types for CDNA5 WMMA
// ---------------------------------------------------------------------------
typedef __attribute__((ext_vector_type(16))) __bf16 v16bf;
typedef __attribute__((ext_vector_type(8)))  float  v8f;
typedef __attribute__((ext_vector_type(4)))  int    v4i;

// fp32 -> bf16 with round-to-nearest-even (manual fallback)
__device__ __forceinline__ unsigned short f2bf_manual(float f) {
    union { float f; unsigned int u; } v; v.f = f;
    unsigned int u = v.u;
    unsigned int r = u + 0x7FFFu + ((u >> 16) & 1u);
    return (unsigned short)(r >> 16);
}

// packed f32x2 -> bf16x2 (uses v_cvt_pk_bf16_f32 when available)
__device__ __forceinline__ unsigned int pack2bf(float x, float y) {
#if __has_builtin(__builtin_amdgcn_cvt_pk_bf16_f32)
    typedef __attribute__((ext_vector_type(2))) __bf16 v2bf;
    v2bf p = __builtin_amdgcn_cvt_pk_bf16_f32(x, y);
    return __builtin_bit_cast(unsigned int, p);
#else
    return ((unsigned int)f2bf_manual(y) << 16) | f2bf_manual(x);
#endif
}

__device__ __forceinline__ unsigned short f2bf(float x) {
#if __has_builtin(__builtin_amdgcn_cvt_pk_bf16_f32)
    return (unsigned short)(pack2bf(x, 0.f) & 0xFFFFu);
#else
    return f2bf_manual(x);
#endif
}

// async global->LDS copy of 16 bytes per lane (gfx1250 ASYNCcnt path).
// Probe-derived signature: (v4i AS1* src, v4i AS3* dst, imm offset, imm cpol)
typedef __attribute__((address_space(1))) v4i* gptr_v4i;
typedef __attribute__((address_space(3))) v4i* lptr_v4i;

__device__ __forceinline__ void async_copy16(const float* g, float* l) {
#if __has_builtin(__builtin_amdgcn_global_load_async_to_lds_b128)
    __builtin_amdgcn_global_load_async_to_lds_b128(
        (gptr_v4i)g, (lptr_v4i)l, 0, 0);
#else
    *reinterpret_cast<float4*>(l) = *reinterpret_cast<const float4*>(g);
#endif
}

__device__ __forceinline__ void wait_async_lds() {
#if __has_builtin(__builtin_amdgcn_s_wait_asynccnt)
    __builtin_amdgcn_s_wait_asynccnt(0);
#else
    asm volatile("s_wait_asynccnt 0x0" ::: "memory");
#endif
}

struct BPair { uint4 a, b; };  // 32 bytes

__device__ __forceinline__ v16bf ld_frag(const unsigned short* p0,
                                         const unsigned short* p1) {
    BPair u;
    u.a = *reinterpret_cast<const uint4*>(p0);
    u.b = *reinterpret_cast<const uint4*>(p1);
    return __builtin_bit_cast(v16bf, u);
}

// ---------------------------------------------------------------------------
// WMMA GEMM:  out = epi(A[MxK] * W[KxN] + bias)
//   epi 0: none    epi 1: relu
//   epi 2: score += sum_rows( tanh(row + bias) . q )   (matrix NOT stored)
// Block: 256 thr (8 waves). Tile: BM=128, BN=64, KT=32. K multiple of 32,
// N multiple of 64 (true for all uses here: N=128).
// ---------------------------------------------------------------------------
__global__ __launch_bounds__(256) void gemm_wmma(
    const float* __restrict__ A, const float* __restrict__ W,
    const float* __restrict__ bias, const float* __restrict__ q,
    float* __restrict__ out, float* __restrict__ score,
    int M, int N, int K, int epi)
{
    __shared__ unsigned short sA[128 * 40];  // bf16 tile, padded stride 40
    __shared__ unsigned short sB[64 * 40];   // bf16 tile, transposed [n][k]

    const int tid  = threadIdx.x;
    const int wv   = tid >> 5;
    const int lane = tid & 31;
    const int half = lane >> 4;
    const int l16  = lane & 15;
    const int m0   = blockIdx.x * 128;
    const int n0   = blockIdx.y * 64;

    v8f acc0 = {}, acc1 = {}, acc2 = {}, acc3 = {};

    for (int k0 = 0; k0 < K; k0 += 32) {
        // ---- stage A: 128x32 f32 -> bf16 (1024 float4 loads across block)
        #pragma unroll
        for (int i = 0; i < 4; ++i) {
            int idx = tid + i * 256;
            int row = idx >> 3;       // 8 float4 per 32-wide row
            int c4  = idx & 7;
            float4 v = make_float4(0.f, 0.f, 0.f, 0.f);
            int gr = m0 + row;
            if (gr < M)
                v = *reinterpret_cast<const float4*>(A + (size_t)gr * K + k0 + c4 * 4);
            unsigned int p0 = pack2bf(v.x, v.y);
            unsigned int p1 = pack2bf(v.z, v.w);
            *reinterpret_cast<uint2*>(&sA[row * 40 + c4 * 4]) = make_uint2(p0, p1);
        }
        // ---- stage B transposed: 32x64 f32 -> bf16 [n][k]
        #pragma unroll
        for (int i = 0; i < 2; ++i) {
            int idx = tid + i * 256;
            int kr = idx >> 4;        // 16 float4 per 64-wide row
            int c4 = idx & 15;
            float4 v = *reinterpret_cast<const float4*>(
                W + (size_t)(k0 + kr) * N + n0 + c4 * 4);
            sB[(c4 * 4 + 0) * 40 + kr] = f2bf(v.x);
            sB[(c4 * 4 + 1) * 40 + kr] = f2bf(v.y);
            sB[(c4 * 4 + 2) * 40 + kr] = f2bf(v.z);
            sB[(c4 * 4 + 3) * 40 + kr] = f2bf(v.w);
        }
        __syncthreads();

        // ---- fragments + WMMA
        // A frag (16x32 bf16): lane(half,m): run K=[half*16..+15] at fixed m
        const unsigned short* aB = &sA[(wv * 16 + l16) * 40];
        v16bf af = ld_frag(aB + half * 16, aB + half * 16 + 8);
        // B frag (32x16 bf16): lane(half,n): run K=[half*16..+15] at fixed n
        {
            const unsigned short* p = &sB[(0 * 16 + l16) * 40 + half * 16];
            v16bf bf = ld_frag(p, p + 8);
            acc0 = __builtin_amdgcn_wmma_f32_16x16x32_bf16(false, af, false, bf,
                                                           (short)0, acc0, false, false);
        }
        {
            const unsigned short* p = &sB[(1 * 16 + l16) * 40 + half * 16];
            v16bf bf = ld_frag(p, p + 8);
            acc1 = __builtin_amdgcn_wmma_f32_16x16x32_bf16(false, af, false, bf,
                                                           (short)0, acc1, false, false);
        }
        {
            const unsigned short* p = &sB[(2 * 16 + l16) * 40 + half * 16];
            v16bf bf = ld_frag(p, p + 8);
            acc2 = __builtin_amdgcn_wmma_f32_16x16x32_bf16(false, af, false, bf,
                                                           (short)0, acc2, false, false);
        }
        {
            const unsigned short* p = &sB[(3 * 16 + l16) * 40 + half * 16];
            v16bf bf = ld_frag(p, p + 8);
            acc3 = __builtin_amdgcn_wmma_f32_16x16x32_bf16(false, af, false, bf,
                                                           (short)0, acc3, false, false);
        }
        __syncthreads();
    }

    // ---- epilogue.  C/D layout: lane(half,n); VGPR r -> row = half*8 + r, col = n
    const int rowbase = m0 + wv * 16 + half * 8;
    if (epi == 2) {
        float lsum = 0.f;
        {
            int col = n0 + 0 * 16 + l16; float bv = bias[col], qv = q[col];
            #pragma unroll
            for (int r = 0; r < 8; ++r)
                if (rowbase + r < M) lsum += tanhf(acc0[r] + bv) * qv;
        }
        {
            int col = n0 + 1 * 16 + l16; float bv = bias[col], qv = q[col];
            #pragma unroll
            for (int r = 0; r < 8; ++r)
                if (rowbase + r < M) lsum += tanhf(acc1[r] + bv) * qv;
        }
        {
            int col = n0 + 2 * 16 + l16; float bv = bias[col], qv = q[col];
            #pragma unroll
            for (int r = 0; r < 8; ++r)
                if (rowbase + r < M) lsum += tanhf(acc2[r] + bv) * qv;
        }
        {
            int col = n0 + 3 * 16 + l16; float bv = bias[col], qv = q[col];
            #pragma unroll
            for (int r = 0; r < 8; ++r)
                if (rowbase + r < M) lsum += tanhf(acc3[r] + bv) * qv;
        }
        // reduce across the 16 lanes of this half (masks < 16 stay in-half)
        #pragma unroll
        for (int off = 8; off; off >>= 1) lsum += __shfl_xor(lsum, off, 32);
        if (l16 == 0) atomicAdd(score, lsum);
    } else {
        {
            int col = n0 + 0 * 16 + l16; float bv = bias[col];
            #pragma unroll
            for (int r = 0; r < 8; ++r) {
                int row = rowbase + r;
                if (row < M) {
                    float v = acc0[r] + bv;
                    if (epi == 1) v = v > 0.f ? v : 0.f;
                    out[(size_t)row * N + col] = v;
                }
            }
        }
        {
            int col = n0 + 1 * 16 + l16; float bv = bias[col];
            #pragma unroll
            for (int r = 0; r < 8; ++r) {
                int row = rowbase + r;
                if (row < M) {
                    float v = acc1[r] + bv;
                    if (epi == 1) v = v > 0.f ? v : 0.f;
                    out[(size_t)row * N + col] = v;
                }
            }
        }
        {
            int col = n0 + 2 * 16 + l16; float bv = bias[col];
            #pragma unroll
            for (int r = 0; r < 8; ++r) {
                int row = rowbase + r;
                if (row < M) {
                    float v = acc2[r] + bv;
                    if (epi == 1) v = v > 0.f ? v : 0.f;
                    out[(size_t)row * N + col] = v;
                }
            }
        }
        {
            int col = n0 + 3 * 16 + l16; float bv = bias[col];
            #pragma unroll
            for (int r = 0; r < 8; ++r) {
                int row = rowbase + r;
                if (row < M) {
                    float v = acc3[r] + bv;
                    if (epi == 1) v = v > 0.f ? v : 0.f;
                    out[(size_t)row * N + col] = v;
                }
            }
        }
    }
}

// ---------------------------------------------------------------------------
// Per-row attention dots (one wave32 per row, H=128 => 4 floats/lane)
// ---------------------------------------------------------------------------
__global__ __launch_bounds__(256) void rowdot2_kernel(
    const float* __restrict__ x, const float* __restrict__ v1,
    const float* __restrict__ v2, float* __restrict__ x1o,
    float* __restrict__ w2o, int Ns)
{
    int wv = threadIdx.x >> 5, lane = threadIdx.x & 31;
    int row = blockIdx.x * 8 + wv;
    if (row >= Ns) return;
    float4 xv = *reinterpret_cast<const float4*>(x + (size_t)row * 128 + lane * 4);
    float4 av = *reinterpret_cast<const float4*>(v1 + lane * 4);
    float4 bv = *reinterpret_cast<const float4*>(v2 + lane * 4);
    float da = xv.x * av.x + xv.y * av.y + xv.z * av.z + xv.w * av.w;
    float db = xv.x * bv.x + xv.y * bv.y + xv.z * bv.z + xv.w * bv.w;
    #pragma unroll
    for (int off = 16; off; off >>= 1) {
        da += __shfl_xor(da, off, 32);
        db += __shfl_xor(db, off, 32);
    }
    if (lane == 0) {
        x1o[row] = da;
        float s = da + db;
        s = s > 0.f ? s : 0.2f * s;       // LeakyReLU(0.2)
        w2o[row] = expf(s);
    }
}

__global__ __launch_bounds__(256) void rowdot1_kernel(
    const float* __restrict__ h, const float* __restrict__ v2,
    float* __restrict__ h1o, int Nt)
{
    int wv = threadIdx.x >> 5, lane = threadIdx.x & 31;
    int row = blockIdx.x * 8 + wv;
    if (row >= Nt) return;
    float4 xv = *reinterpret_cast<const float4*>(h + (size_t)row * 128 + lane * 4);
    float4 bv = *reinterpret_cast<const float4*>(v2 + lane * 4);
    float d = xv.x * bv.x + xv.y * bv.y + xv.z * bv.z + xv.w * bv.w;
    #pragma unroll
    for (int off = 16; off; off >>= 1) d += __shfl_xor(d, off, 32);
    if (lane == 0) h1o[row] = d;
}

// agg = w2*x ; div = w2   (self-loop init, overwrites poisoned scratch)
__global__ __launch_bounds__(256) void gat_init_kernel(
    float* __restrict__ agg, float* __restrict__ divb,
    const float* __restrict__ x, const float* __restrict__ w2, long long total)
{
    long long idx = (long long)blockIdx.x * 256 + threadIdx.x;
    if (idx >= total) return;
    long long row = idx >> 7;
    float w = w2[row];
    agg[idx] = w * x[idx];
    if ((idx & 127) == 0) divb[row] = w;
}

// one wave per edge; 51MB accumulator fits MI455X's 192MB L2 -> atomics stay on-chip
__global__ __launch_bounds__(256) void gat_edges_kernel(
    const int* __restrict__ s, const int* __restrict__ t,
    const float* __restrict__ x1, const float* __restrict__ h1,
    const float* __restrict__ h, float* __restrict__ agg,
    float* __restrict__ divb, int E)
{
    int wv = threadIdx.x >> 5, lane = threadIdx.x & 31;
    int e = blockIdx.x * 8 + wv;
    if (e >= E) return;
    int si = s[e], ti = t[e];
    float v = x1[si] + h1[ti];
    v = v > 0.f ? v : 0.2f * v;
    float w = expf(v);
    float4 hv = *reinterpret_cast<const float4*>(h + (size_t)ti * 128 + lane * 4);
    float* ap = agg + (size_t)si * 128 + lane * 4;
    atomicAdd(ap + 0, w * hv.x);
    atomicAdd(ap + 1, w * hv.y);
    atomicAdd(ap + 2, w * hv.z);
    atomicAdd(ap + 3, w * hv.w);
    if (lane == 0) atomicAdd(divb + si, w);
}

__global__ __launch_bounds__(256) void gat_norm_kernel(
    float* __restrict__ agg, const float* __restrict__ divb, long long total)
{
    long long idx = (long long)blockIdx.x * 256 + threadIdx.x;
    if (idx >= total) return;
    agg[idx] /= divb[idx >> 7];
}

__global__ void zero2_kernel(float* s) { if (threadIdx.x < 2) s[threadIdx.x] = 0.f; }

// paper semantic combine: beta = softmax(score/N); out = elu(b0*pa + b1*pp)
__global__ __launch_bounds__(256) void sem_combine2_kernel(
    const float* __restrict__ pa, const float* __restrict__ pp,
    const float* __restrict__ sc, float* __restrict__ outp,
    float invN, long long total)
{
    long long idx = (long long)blockIdx.x * 256 + threadIdx.x;
    if (idx >= total) return;
    float s0 = sc[0] * invN, s1 = sc[1] * invN;
    float m = fmaxf(s0, s1);
    float e0 = expf(s0 - m), e1 = expf(s1 - m);
    float inv = 1.f / (e0 + e1);
    float v = (e0 * inv) * pa[idx] + (e1 * inv) * pp[idx];
    outp[idx] = v > 0.f ? v : expm1f(v);
}

// author branch: M=1 => softmax == 1 => just ELU (exact)
__global__ __launch_bounds__(256) void elu_inplace_kernel(
    float* __restrict__ b, long long total)
{
    long long idx = (long long)blockIdx.x * 256 + threadIdx.x;
    if (idx >= total) return;
    float v = b[idx];
    b[idx] = v > 0.f ? v : expm1f(v);
}

// fc2: [M,128] x [128,16] + bias; W staged to LDS via async global->LDS DMA
__global__ __launch_bounds__(256) void fc2_kernel(
    const float* __restrict__ x, const float* __restrict__ W,
    const float* __restrict__ b, float* __restrict__ out, int M)
{
    __shared__ float sW[128 * 16];
    // 8KB block-uniform weight tile: 512 x 16B lane-copies, 2 per thread
    #pragma unroll
    for (int i = 0; i < 2; ++i) {
        int idx = threadIdx.x + i * 256;          // float4 index
        async_copy16(W + (size_t)idx * 4, &sW[idx * 4]);
    }
    wait_async_lds();
    __syncthreads();

    int row = blockIdx.x * 16 + (threadIdx.x >> 4);
    int o = threadIdx.x & 15;
    if (row >= M) return;
    const float* xr = x + (size_t)row * 128;
    float sum = 0.f;
    #pragma unroll 4
    for (int k = 0; k < 128; ++k) sum += xr[k] * sW[k * 16 + o];
    out[(size_t)row * 16 + o] = sum + b[o];
}

// ---------------------------------------------------------------------------
extern "C" void kernel_launch(void* const* d_in, const int* in_sizes, int n_in,
                              void* d_out, int out_size, void* d_ws, size_t ws_size,
                              hipStream_t stream)
{
    (void)n_in; (void)out_size; (void)ws_size;
    const float* x_paper  = (const float*)d_in[0];
    const float* x_author = (const float*)d_in[1];
    const int*   ei_pa    = (const int*)d_in[2];
    const int*   ei_ap    = (const int*)d_in[3];
    const int*   ei_pp    = (const int*)d_in[4];
    const float* fc1pw    = (const float*)d_in[5];
    const float* fc1pb    = (const float*)d_in[6];
    const float* fc1aw    = (const float*)d_in[7];
    const float* fc1ab    = (const float*)d_in[8];
    const float* fcsw     = (const float*)d_in[9];
    const float* fcsb     = (const float*)d_in[10];
    const float* a1       = (const float*)d_in[11];
    const float* a2       = (const float*)d_in[12];
    const float* semw     = (const float*)d_in[13];
    const float* semb     = (const float*)d_in[14];
    const float* semq     = (const float*)d_in[15];
    const float* fc2w     = (const float*)d_in[16];
    const float* fc2b     = (const float*)d_in[17];
    float* out = (float*)d_out;

    const int H = 128, DIN = 256;
    const int NP = in_sizes[0] / DIN;
    const int NA = in_sizes[1] / DIN;
    const int E  = in_sizes[2] / 2;
    const size_t NPH = (size_t)NP * H, NAH = (size_t)NA * H;
    const size_t NMAX = (NP > NA ? (size_t)NP : (size_t)NA);

    float* p = (float*)d_ws;
    float* xp   = p; p += NPH;    // also hosts 'pa'
    float* xa   = p; p += NAH;    // also hosts 'ap'
    float* xp2  = p; p += NPH;
    float* xa2  = p; p += NAH;
    float* ppb  = p; p += NPH;
    float* x1b  = p; p += NMAX + 64;
    float* h1b  = p; p += NMAX + 64;
    float* w2b  = p; p += NMAX + 64;
    float* divb = p; p += NMAX + 64;
    float* score = p;

    auto gemm = [&](const float* A, const float* W, const float* B, const float* Q,
                    float* O, float* S, int M, int N, int K, int epi) {
        dim3 g((unsigned)((M + 127) / 128), (unsigned)(N / 64));
        gemm_wmma<<<g, 256, 0, stream>>>(A, W, B, Q, O, S, M, N, K, epi);
    };
    auto run_gat = [&](const float* x, int Ns, const float* h, int Nt,
                       const int* sI, const int* tI,
                       const float* v1, const float* v2, float* ob) {
        rowdot2_kernel<<<dim3((Ns + 7) / 8), 256, 0, stream>>>(x, v1, v2, x1b, w2b, Ns);
        rowdot1_kernel<<<dim3((Nt + 7) / 8), 256, 0, stream>>>(h, v2, h1b, Nt);
        long long tot = (long long)Ns * H;
        gat_init_kernel<<<dim3((unsigned)((tot + 255) / 256)), 256, 0, stream>>>(
            ob, divb, x, w2b, tot);
        gat_edges_kernel<<<dim3((E + 7) / 8), 256, 0, stream>>>(
            sI, tI, x1b, h1b, h, ob, divb, E);
        gat_norm_kernel<<<dim3((unsigned)((tot + 255) / 256)), 256, 0, stream>>>(
            ob, divb, tot);
    };

    // fc1 + ReLU
    gemm(x_paper,  fc1pw, fc1pb, nullptr, xp, nullptr, NP, H, DIN, 1);
    gemm(x_author, fc1aw, fc1ab, nullptr, xa, nullptr, NA, H, DIN, 1);

    for (int i = 0; i < 2; ++i) {
        const float* wfc = fcsw + (size_t)i * H * H;
        const float* bfc = fcsb + (size_t)i * H;
        gemm(xp, wfc, bfc, nullptr, xp2, nullptr, NP, H, H, 0);
        gemm(xa, wfc, bfc, nullptr, xa2, nullptr, NA, H, H, 0);

        const float* v1_pa = a1 + (size_t)(i * 3 + 0) * H;
        const float* v2_pa = a2 + (size_t)(i * 3 + 0) * H;
        const float* v1_ap = a1 + (size_t)(i * 3 + 1) * H;
        const float* v2_ap = a2 + (size_t)(i * 3 + 1) * H;
        const float* v1_pp = a1 + (size_t)(i * 3 + 2) * H;
        const float* v2_pp = a2 + (size_t)(i * 3 + 2) * H;

        run_gat(xp2, NP, xa2, NA, ei_pa, ei_pa + E, v1_pa, v2_pa, xp);   // pa -> xp
        run_gat(xa2, NA, xp2, NP, ei_ap, ei_ap + E, v1_ap, v2_ap, xa);   // ap -> xa
        run_gat(xp2, NP, xp2, NP, ei_pp, ei_pp + E, v1_pp, v2_pp, ppb);  // pp

        // paper semantic attention: logits via fused tanh.q WMMA epilogue
        const float* swp = semw + (size_t)(i * 2 + 0) * H * H;
        const float* sbp = semb + (size_t)(i * 2 + 0) * H;
        const float* sqp = semq + (size_t)(i * 2 + 0) * H;
        zero2_kernel<<<1, 32, 0, stream>>>(score);
        gemm(xp,  swp, sbp, sqp, nullptr, score + 0, NP, H, H, 2);
        gemm(ppb, swp, sbp, sqp, nullptr, score + 1, NP, H, H, 2);
        long long totp = (long long)NP * H;
        sem_combine2_kernel<<<dim3((unsigned)((totp + 255) / 256)), 256, 0, stream>>>(
            xp, ppb, score, xp, 1.f / (float)NP, totp);

        // author branch: single metapath -> ELU only
        long long tota = (long long)NA * H;
        elu_inplace_kernel<<<dim3((unsigned)((tota + 255) / 256)), 256, 0, stream>>>(
            xa, tota);
    }

    fc2_kernel<<<dim3((NP + 15) / 16), 256, 0, stream>>>(xp, fc2w, fc2b, out, NP);
}
